// MultiHeadSelfAttention_43619687858448
// MI455X (gfx1250) — matrix operand order repaired and checked by reading, not compiled, not run on hardware
//
#include <hip/hip_runtime.h>
#include <hip/hip_bf16.h>

typedef _Float16 f16;
typedef _Float16 h4  __attribute__((ext_vector_type(4)));
typedef _Float16 h8  __attribute__((ext_vector_type(8)));
typedef _Float16 h16 __attribute__((ext_vector_type(16)));
typedef float    f8  __attribute__((ext_vector_type(8)));

#define B_   4
#define N_   2048
#define D_   1024
#define H_   16
#define HD_  64

// ---------------- helpers ----------------

static __device__ inline h16 cat8(h8 a, h8 b) {
  h16 r;
#pragma unroll
  for (int i = 0; i < 8; ++i) { r[i] = a[i]; r[i + 8] = b[i]; }
  return r;
}

// A-fragment (16xK=32, f16) from row-major src[ld]: rows m0..m0+15, cols kb..kb+31.
// ISA layout: lanes 0-15 hold row m, K{0..7,16..23}; lanes 16-31 hold K{8..15,24..31}.
static __device__ inline h16 load_afrag(const f16* __restrict__ src, int ld, int m0, int kb) {
  const int lane = threadIdx.x & 31;
  const int r  = m0 + (lane & 15);
  const int ks = kb + ((lane >> 4) << 3);
  const f16* p = src + (size_t)r * ld + ks;
  h8 lo = *(const h8*)(p);
  h8 hi = *(const h8*)(p + 16);
  return cat8(lo, hi);
}

// B-fragment (K=32 x 16, f16): B(k,c) = W[n0+c][kb+k] (i.e. W rows are the B columns).
// ISA layout: lanes 0-15 hold col c, K{0..7,16..23}; lanes 16-31 hold K{8..15,24..31}.
static __device__ inline h16 load_bfrag(const f16* __restrict__ W, int ld, int n0, int kb) {
  const int lane = threadIdx.x & 31;
  const int c  = n0 + (lane & 15);
  const int ks = kb + ((lane >> 4) << 3);
  const f16* p = W + (size_t)c * ld + ks;
  h8 lo = *(const h8*)(p);
  h8 hi = *(const h8*)(p + 16);
  return cat8(lo, hi);
}

static __device__ inline f8 wmma(h16 a, h16 b, f8 c) {
  return __builtin_amdgcn_wmma_f32_16x16x32_f16(
      /*neg_a=*/false, a, /*neg_b=*/false, b,
      /*c_mod=*/(short)0, c, /*reuse_a=*/false, /*reuse_b=*/false);
}

// CDNA5 async global -> LDS, GVS mode: LDS[lds_off] = MEM[sbase + goff], 16 bytes.
// Tracked by ASYNCcnt (no VGPR landing).
static __device__ inline void async_b128(unsigned lds_off, const void* sbase, unsigned goff) {
  asm volatile("global_load_async_to_lds_b128 %0, %1, %2"
               :: "v"(lds_off), "v"(goff), "s"(sbase)
               : "memory");
}

// ---------------- fp32 -> fp16 convert ----------------

__global__ __launch_bounds__(256) void cvt_f32_f16(const float* __restrict__ s,
                                                   f16* __restrict__ d, long n4) {
  long i = (long)blockIdx.x * blockDim.x + threadIdx.x;
  if (i < n4) {
    float4 v = ((const float4*)s)[i];
    h4 o; o[0] = (f16)v.x; o[1] = (f16)v.y; o[2] = (f16)v.z; o[3] = (f16)v.w;
    ((h4*)d)[i] = o;
  }
}

// ---------------- QKV projection: Y = (X @ W^T + bias) * scale ----------------
// mode 1: store [B,H,N,Hd]  (for Q and K)
// mode 2: store [B,H,Hd,N]  (transposed V for the P*V B-fragments)
__global__ __launch_bounds__(256) void proj_f16(const f16* __restrict__ X,
                                                const f16* __restrict__ W,
                                                const float* __restrict__ bias,
                                                f16* __restrict__ Y,
                                                float scale, int mode) {
  const int lane = threadIdx.x & 31;
  const int wave = threadIdx.x >> 5;
  const int wm = wave & 1;   // 2 waves along M
  const int wn = wave >> 1;  // 4 waves along N
  const int rowBase = blockIdx.x * 128 + wm * 64;  // 4 M-tiles of 16
  const int colBase = blockIdx.y * 128 + wn * 32;  // 2 N-tiles of 16

  f8 acc[4][2];
#pragma unroll
  for (int i = 0; i < 4; ++i)
#pragma unroll
    for (int t = 0; t < 2; ++t) acc[i][t] = 0.0f;

  for (int kb = 0; kb < D_; kb += 32) {
    h16 a[4], bb[2];
#pragma unroll
    for (int i = 0; i < 4; ++i) a[i] = load_afrag(X, D_, rowBase + i * 16, kb);
#pragma unroll
    for (int t = 0; t < 2; ++t) bb[t] = load_bfrag(W, D_, colBase + t * 16, kb);
#pragma unroll
    for (int i = 0; i < 4; ++i)
#pragma unroll
      for (int t = 0; t < 2; ++t) acc[i][t] = wmma(a[i], bb[t], acc[i][t]);
  }

  const int hi = (lane >> 4) << 3;
  const int cl = lane & 15;
#pragma unroll
  for (int i = 0; i < 4; ++i) {
#pragma unroll
    for (int t = 0; t < 2; ++t) {
      const int col = colBase + t * 16 + cl;
      const float bv = bias[col];
      const int h = col >> 6, dd = col & 63;
#pragma unroll
      for (int j = 0; j < 8; ++j) {
        const int row = rowBase + i * 16 + j + hi;      // 0..B*N-1
        const int b = row >> 11, nn = row & (N_ - 1);
        const float v = (acc[i][t][j] + bv) * scale;
        size_t off;
        if (mode == 1)
          off = ((((size_t)b * H_ + h) * N_ + nn) * HD_ + dd);
        else
          off = ((((size_t)b * H_ + h) * HD_ + dd) * N_ + nn);
        Y[off] = (f16)v;
      }
    }
  }
}

// ---------------- flash attention with async K/V staging ----------------
// Q,K: [B,H,N,Hd] f16 (Q pre-scaled by Hd^-0.5), VT: [B,H,Hd,N] f16.
// Output AO: [B,N,D] f16 (heads re-interleaved).
__global__ __launch_bounds__(256) void attn_flash(const f16* __restrict__ Q,
                                                  const f16* __restrict__ K,
                                                  const f16* __restrict__ VT,
                                                  f16* __restrict__ AO) {
  __shared__ __attribute__((aligned(16))) f16 kt[2][32 * 64];  // 32 keys x 64 d (row-major)
  __shared__ __attribute__((aligned(16))) f16 vt[2][64 * 32];  // 64 d x 32 keys (row-major)
  __shared__ __attribute__((aligned(16))) f16 pt[8][16 * 32];  // per-wave P tile

  const int tid  = threadIdx.x;
  const int lane = tid & 31;
  const int wave = tid >> 5;
  const int bh    = blockIdx.x >> 4;   // b*H + h
  const int chunk = blockIdx.x & 15;
  const int b = bh >> 4, h = bh & 15;
  const int q0 = chunk * 128 + wave * 16;

  const f16* qp = Q  + (size_t)bh * N_ * HD_;
  const f16* kp = K  + (size_t)bh * N_ * HD_;
  const f16* vp = VT + (size_t)bh * HD_ * N_;
  f16* myp = &pt[wave][0];

  // staging addresses: both K and VT tiles are exactly tid*16 bytes into the tile
  const unsigned kLds0 = (unsigned)(uintptr_t)&kt[0][0] + (unsigned)tid * 16;
  const unsigned kLds1 = (unsigned)(uintptr_t)&kt[1][0] + (unsigned)tid * 16;
  const unsigned vLds0 = (unsigned)(uintptr_t)&vt[0][0] + (unsigned)tid * 16;
  const unsigned vLds1 = (unsigned)(uintptr_t)&vt[1][0] + (unsigned)tid * 16;
  const unsigned kG = (unsigned)tid * 16;                                  // + n0*128
  const unsigned vG = (((unsigned)(tid >> 2) * N_ + (unsigned)(tid & 3) * 8)) * 2;  // + n0*2

  const h16 qa0 = load_afrag(qp, HD_, q0, 0);
  const h16 qa1 = load_afrag(qp, HD_, q0, 32);

  f8 o[4];
#pragma unroll
  for (int t = 0; t < 4; ++t) o[t] = 0.0f;
  float mrun[8], lrun[8];
#pragma unroll
  for (int j = 0; j < 8; ++j) { mrun[j] = -1e30f; lrun[j] = 0.0f; }

  // prologue: stage first K/V block into buffer 0
  async_b128(kLds0, kp, kG);
  async_b128(vLds0, vp, vG);

  int buf = 0;
  for (int n0 = 0; n0 < N_; n0 += 32) {
    // stage next block into the other buffer, wait for current block to land
    if (n0 + 32 < N_) {
      const unsigned kl = buf ? kLds0 : kLds1;
      const unsigned vl = buf ? vLds0 : vLds1;
      async_b128(kl, kp, kG + (unsigned)(n0 + 32) * 128);
      async_b128(vl, vp, vG + (unsigned)(n0 + 32) * 2);
      asm volatile("s_wait_asynccnt 2" ::: "memory");  // old 2 done, new 2 in flight
    } else {
      asm volatile("s_wait_asynccnt 0" ::: "memory");
    }
    __syncthreads();  // all threads' staging of `buf` visible to all waves

    const f16* ktp = &kt[buf][0];
    const f16* vtp = &vt[buf][0];

    // S = Q * K^T for 32 keys (two 16-col tiles), K-frags from LDS
    f8 s0 = 0.0f, s1 = 0.0f;
    h16 b00 = load_bfrag(ktp, HD_, 0,  0);
    h16 b01 = load_bfrag(ktp, HD_, 0,  32);
    h16 b10 = load_bfrag(ktp, HD_, 16, 0);
    h16 b11 = load_bfrag(ktp, HD_, 16, 32);
    s0 = wmma(qa0, b00, s0); s0 = wmma(qa1, b01, s0);
    s1 = wmma(qa0, b10, s1); s1 = wmma(qa1, b11, s1);

    // online softmax; row j (lanes<16) / j+8 (lanes>=16), 16 cols per 16-lane group
#pragma unroll
    for (int j = 0; j < 8; ++j) {
      float mx = fmaxf(s0[j], s1[j]);
      mx = fmaxf(mx, __shfl_xor(mx, 1));
      mx = fmaxf(mx, __shfl_xor(mx, 2));
      mx = fmaxf(mx, __shfl_xor(mx, 4));
      mx = fmaxf(mx, __shfl_xor(mx, 8));
      const float mnew = fmaxf(mrun[j], mx);
      const float corr = __expf(mrun[j] - mnew);
      mrun[j] = mnew;
      const float p0 = __expf(s0[j] - mnew);
      const float p1 = __expf(s1[j] - mnew);
      float rs = p0 + p1;
      rs += __shfl_xor(rs, 1);
      rs += __shfl_xor(rs, 2);
      rs += __shfl_xor(rs, 4);
      rs += __shfl_xor(rs, 8);
      lrun[j] = lrun[j] * corr + rs;
      const int row = j + ((lane >> 4) << 3);
      const int cc  = lane & 15;
      myp[row * 32 + cc]      = (f16)p0;   // C-layout -> row-major P tile in LDS
      myp[row * 32 + 16 + cc] = (f16)p1;
      o[0][j] *= corr; o[1][j] *= corr; o[2][j] *= corr; o[3][j] *= corr;
    }

    asm volatile("s_wait_dscnt 0" ::: "memory");  // our P stores visible to our loads

    // O += P(16x32) * V(32x64): P as A-frag, V^T rows are B columns (from LDS)
    const h16 pa = load_afrag(myp, 32, 0, 0);
#pragma unroll
    for (int t = 0; t < 4; ++t) {
      h16 vb = load_bfrag(vtp, 32, t * 16, 0);
      o[t] = wmma(pa, vb, o[t]);
    }

    __syncthreads();  // everyone done reading `buf` before it is restaged
    buf ^= 1;
  }

  const int hi = (lane >> 4) << 3;
  const int cl = lane & 15;
#pragma unroll
  for (int j = 0; j < 8; ++j) {
    const float inv = 1.0f / lrun[j];
    const int row = q0 + j + hi;
    const size_t base = ((size_t)b * N_ + row) * D_ + h * HD_;
#pragma unroll
    for (int t = 0; t < 4; ++t)
      AO[base + t * 16 + cl] = (f16)(o[t][j] * inv);
  }
}

// ---------------- output projection: out = AO @ Wo^T + bo (fp32 out) ----------------
__global__ __launch_bounds__(256) void proj_f32_out(const f16* __restrict__ X,
                                                    const f16* __restrict__ W,
                                                    const float* __restrict__ bias,
                                                    float* __restrict__ Y) {
  const int lane = threadIdx.x & 31;
  const int wave = threadIdx.x >> 5;
  const int wm = wave & 1;
  const int wn = wave >> 1;
  const int rowBase = blockIdx.x * 128 + wm * 64;
  const int colBase = blockIdx.y * 128 + wn * 32;

  f8 acc[4][2];
#pragma unroll
  for (int i = 0; i < 4; ++i)
#pragma unroll
    for (int t = 0; t < 2; ++t) acc[i][t] = 0.0f;

  for (int kb = 0; kb < D_; kb += 32) {
    h16 a[4], bb[2];
#pragma unroll
    for (int i = 0; i < 4; ++i) a[i] = load_afrag(X, D_, rowBase + i * 16, kb);
#pragma unroll
    for (int t = 0; t < 2; ++t) bb[t] = load_bfrag(W, D_, colBase + t * 16, kb);
#pragma unroll
    for (int i = 0; i < 4; ++i)
#pragma unroll
      for (int t = 0; t < 2; ++t) acc[i][t] = wmma(a[i], bb[t], acc[i][t]);
  }

  const int hi = (lane >> 4) << 3;
  const int cl = lane & 15;
#pragma unroll
  for (int i = 0; i < 4; ++i)
#pragma unroll
    for (int t = 0; t < 2; ++t) {
      const int col = colBase + t * 16 + cl;
      const float bv = bias[col];
#pragma unroll
      for (int j = 0; j < 8; ++j) {
        const int row = rowBase + i * 16 + j + hi;
        Y[(size_t)row * D_ + col] = acc[i][t][j] + bv;
      }
    }
}

// ---------------- launch ----------------

extern "C" void kernel_launch(void* const* d_in, const int* in_sizes, int n_in,
                              void* d_out, int out_size, void* d_ws, size_t ws_size,
                              hipStream_t stream) {
  const float* x  = (const float*)d_in[0];
  const float* Wq = (const float*)d_in[1];
  const float* bq = (const float*)d_in[2];
  const float* Wk = (const float*)d_in[3];
  const float* bk = (const float*)d_in[4];
  const float* Wv = (const float*)d_in[5];
  const float* bv = (const float*)d_in[6];
  const float* Wo = (const float*)d_in[7];
  const float* bo = (const float*)d_in[8];
  float* out = (float*)d_out;

  const size_t BND = (size_t)B_ * N_ * D_;  // 8M elems
  const size_t DD  = (size_t)D_ * D_;       // 1M elems

  f16* xh  = (f16*)d_ws;
  f16* wqh = xh  + BND;
  f16* wkh = wqh + DD;
  f16* wvh = wkh + DD;
  f16* woh = wvh + DD;
  f16* qb  = woh + DD;   // [B,H,N,Hd], pre-scaled
  f16* kb  = qb  + BND;  // [B,H,N,Hd]
  f16* vT  = kb  + BND;  // [B,H,Hd,N]
  f16* ao  = vT  + BND;  // [B,N,D]

  // fp32 -> fp16
  {
    long n4 = (long)(BND / 4);
    cvt_f32_f16<<<(unsigned)((n4 + 255) / 256), 256, 0, stream>>>(x, xh, n4);
    long w4 = (long)(DD / 4);
    unsigned wbk = (unsigned)((w4 + 255) / 256);
    cvt_f32_f16<<<wbk, 256, 0, stream>>>(Wq, wqh, w4);
    cvt_f32_f16<<<wbk, 256, 0, stream>>>(Wk, wkh, w4);
    cvt_f32_f16<<<wbk, 256, 0, stream>>>(Wv, wvh, w4);
    cvt_f32_f16<<<wbk, 256, 0, stream>>>(Wo, woh, w4);
  }

  // QKV projections (scale 1/sqrt(Hd) folded into Q)
  dim3 pgrid((B_ * N_) / 128, D_ / 128);
  proj_f16<<<pgrid, 256, 0, stream>>>(xh, wqh, bq, qb, 0.125f, 1);
  proj_f16<<<pgrid, 256, 0, stream>>>(xh, wkh, bk, kb, 1.0f,   1);
  proj_f16<<<pgrid, 256, 0, stream>>>(xh, wvh, bv, vT, 1.0f,   2);

  // flash attention: 64 (b,h) pairs x 16 chunks of 128 query rows
  attn_flash<<<B_ * H_ * (N_ / 128), 256, 0, stream>>>(qb, kb, vT, ao);

  // output projection (fp32 + bias) straight into d_out
  proj_f32_out<<<pgrid, 256, 0, stream>>>(ao, woh, bo, out);
}
